// MultiHeadAttention_29042568855979
// MI455X (gfx1250) — compile-verified, matrix operand order checked by reading, not run on hardware
//
#include <hip/hip_runtime.h>
#include <hip/hip_bf16.h>

// MHA forward for MI455X (gfx1250, wave32, WMMA + async global->LDS).
// B=4, S=2048, D=1024, H=16, DK=64.

typedef __bf16 bf16_t;
typedef __bf16 v16bf __attribute__((ext_vector_type(16)));
typedef __bf16 v8bf  __attribute__((ext_vector_type(8)));
typedef float  v8f   __attribute__((ext_vector_type(8)));
typedef int    v4i   __attribute__((ext_vector_type(4)));
typedef __attribute__((address_space(1))) v4i gas_v4i;   // global int4
typedef __attribute__((address_space(3))) v4i las_v4i;   // LDS int4

static constexpr int Bsz = 4;
static constexpr int S   = 2048;
static constexpr int D   = 1024;
static constexpr int H   = 16;
static constexpr int DK  = 64;
static constexpr int M   = Bsz * S;   // 8192 rows for all GEMMs

#define DEVI __device__ __forceinline__

#if defined(__HIP_DEVICE_COMPILE__) && \
    __has_builtin(__builtin_amdgcn_global_load_async_to_lds_b128) && \
    __has_builtin(__builtin_amdgcn_s_wait_asynccnt)
#define HAVE_ASYNC_LDS 1
#else
#define HAVE_ASYNC_LDS 0
#endif

DEVI bf16_t f2bf(float f) {
  union { float f; unsigned u; } x; x.f = f;
  unsigned r = x.u + 0x7FFFu + ((x.u >> 16) & 1u);   // round-to-nearest-even
  unsigned short h = (unsigned short)(r >> 16);
  return __builtin_bit_cast(bf16_t, h);
}

// Fragment gather shared by A-layout (lane=row, elems along K) and
// B-layout (lane=col, elems along K): elems 0-7 at p, elems 8-15 at p+16.
DEVI v16bf load_frag(const bf16_t* p) {
  v8bf lo = *(const v8bf*)(p);
  v8bf hi = *(const v8bf*)(p + 16);
  return __builtin_shufflevector(lo, hi,
      0, 1, 2, 3, 4, 5, 6, 7, 8, 9, 10, 11, 12, 13, 14, 15);
}

DEVI v8f wmma_bf16(v16bf a, v16bf b, v8f c) {
  return __builtin_amdgcn_wmma_f32_16x16x32_bf16(
      false, a, false, b, (short)0, c, false, false);
}

// 16-byte global -> LDS copy. Async path uses the CDNA5 async-DMA engine
// (tracked by ASYNCcnt); fallback is a plain load + ds_store.
// Flat->LDS address truncation to 32 bits is the ISA aperture mapping.
DEVI void copy16_g2l(bf16_t* l, const bf16_t* g) {
#if HAVE_ASYNC_LDS
  __builtin_amdgcn_global_load_async_to_lds_b128(
      (gas_v4i*)(unsigned long long)g,
      (las_v4i*)(unsigned)(unsigned long long)l,
      0, 0);
#else
  *(v8bf*)l = *(const v8bf*)g;
#endif
}

DEVI void wait_async() {
#if HAVE_ASYNC_LDS
  __builtin_amdgcn_s_wait_asynccnt(0);
#endif
}

// ---------------------------------------------------------------- convert
__global__ __launch_bounds__(256)
void cvt_f32_bf16(const float* __restrict__ in, bf16_t* __restrict__ out, int n) {
  int i = blockIdx.x * blockDim.x + threadIdx.x;
  if (i < n) out[i] = f2bf(in[i]);
}

// ---------------------------------------------------------------- GEMM Y = X * W^T
// X: (M, 1024) bf16 row-major. W: (1024, 1024) bf16 row-major. Y: (M,1024) f32.
// Block = 8 waves: 2 in M x 4 in N; block tile 64M x 256N; wave tile 32M x 64N.
// Double-buffered LDS staging filled by async global->LDS copies; compute
// reads fragments with ds_load_b128 (row stride padded to 40 elems: the 16
// lanes of a fragment gather land on 16 distinct banks).
__global__ __launch_bounds__(256)
void gemm_xwT(const bf16_t* __restrict__ A, const bf16_t* __restrict__ W,
              float* __restrict__ Y) {
  constexpr int KD  = 1024;
  constexpr int ND  = 1024;
  constexpr int LDA = 40;                    // padded LDS row stride (elems)
  __shared__ bf16_t lA[2][64 * LDA];         // 2 x 5,120 B
  __shared__ bf16_t lW[2][256 * LDA];        // 2 x 20,480 B

  const int tid  = threadIdx.x;
  const int lane = tid & 31;
  const int wid  = tid >> 5;
  const int half = lane >> 4;
  const int lr   = lane & 15;
  const int wm   = wid & 1;                  // 2 M-waves
  const int wn   = wid >> 1;                 // 4 N-waves
  const int m_blk = blockIdx.y * 64;
  const int n_blk = blockIdx.x * 256;

  // cooperative loader: one b128 chunk (8 bf16) per slot
  const int arow  = tid >> 2;                // 0..63
  const int akoff = (tid & 3) * 8;           // 0,8,16,24
  const bf16_t* gA = A + (size_t)(m_blk + arow) * KD + akoff;
  bf16_t* sA[2] = { &lA[0][arow * LDA + akoff], &lA[1][arow * LDA + akoff] };

  const bf16_t* gW[4];
  bf16_t* sW[2][4];
#pragma unroll
  for (int i = 0; i < 4; ++i) {
    int c     = tid + 256 * i;
    int wrow  = c >> 2;                      // 0..255
    int wkoff = (c & 3) * 8;
    gW[i]     = W + (size_t)(n_blk + wrow) * KD + wkoff;
    sW[0][i]  = &lW[0][wrow * LDA + wkoff];
    sW[1][i]  = &lW[1][wrow * LDA + wkoff];
  }

  auto load_stage = [&](int s, int k) {
    copy16_g2l(sA[s], gA + k);
#pragma unroll
    for (int i = 0; i < 4; ++i) copy16_g2l(sW[s][i], gW[i] + k);
  };

  v8f acc[2][4] = {};
  auto compute = [&](int s) {
    const bf16_t* la0 = &lA[s][(wm * 32 + lr) * LDA + half * 8];
    const bf16_t* la1 = la0 + 16 * LDA;
    const bf16_t* lw  = &lW[s][(wn * 64 + lr) * LDA + half * 8];
    v16bf a0 = load_frag(la0);
    v16bf a1 = load_frag(la1);
#pragma unroll
    for (int g = 0; g < 4; ++g) {
      v16bf b = load_frag(lw + g * 16 * LDA);
      acc[0][g] = wmma_bf16(a0, b, acc[0][g]);
      acc[1][g] = wmma_bf16(a1, b, acc[1][g]);
    }
  };

  // pipeline: prime stage 0, then load(next) / compute(cur) / sync / flip
  load_stage(0, 0);
  wait_async();
  __syncthreads();

  int buf = 0;
  for (int k = 32; k < KD; k += 32) {        // 31 iterations, uniform barriers
    load_stage(buf ^ 1, k);
    compute(buf);
    wait_async();
    __syncthreads();
    buf ^= 1;
  }
  compute(buf);

#pragma unroll
  for (int t = 0; t < 2; ++t) {
#pragma unroll
    for (int g = 0; g < 4; ++g) {
      int col = n_blk + wn * 64 + g * 16 + lr;
#pragma unroll
      for (int j = 0; j < 8; ++j) {
        int row = m_blk + wm * 32 + t * 16 + j + 8 * half;   // C/D layout
        Y[(size_t)row * ND + col] = acc[t][g][j];
      }
    }
  }
}

// ---------------------------------------------------------------- RoPE + cvt
// in: (B*S, H*DK) f32 ; out: same layout bf16, rotated pairwise per head.
__global__ __launch_bounds__(256)
void rope_cvt(const float* __restrict__ in, bf16_t* __restrict__ out) {
  int idx = blockIdx.x * blockDim.x + threadIdx.x;   // over B*S*H*(DK/2)
  if (idx >= M * H * (DK / 2)) return;
  int i   = idx & (DK / 2 - 1);       // pair index 0..31
  int h   = (idx >> 5) & (H - 1);
  int row = idx >> 9;                 // b*S + s
  int s   = row & (S - 1);
  float freq = __powf(10000.0f, -(float)(2 * i) / (float)DK);
  float ang  = (float)s * freq;
  float sn, cs;
  __sincosf(ang, &sn, &cs);
  size_t base = (size_t)row * D + h * DK + 2 * i;
  float x1 = in[base], x2 = in[base + 1];
  out[base]     = f2bf(x1 * cs - x2 * sn);
  out[base + 1] = f2bf(x1 * sn + x2 * cs);
}

// ---------------------------------------------------------------- V transpose
// in: (B*S, H*DK) f32 -> out: (B,H,DK,S) bf16
__global__ __launch_bounds__(256)
void vtrans(const float* __restrict__ in, bf16_t* __restrict__ out) {
  int idx = blockIdx.x * blockDim.x + threadIdx.x;
  if (idx >= M * D) return;
  int dk  = idx & (DK - 1);
  int h   = (idx >> 6) & (H - 1);
  int row = idx >> 10;                // b*S + s
  int s   = row & (S - 1);
  int b   = row >> 11;
  out[(((size_t)(b * H + h)) * DK + dk) * S + s] = f2bf(in[idx]);
}

// ---------------------------------------------------------------- flash attention
// One wave per (b, h, 16-query tile). Online softmax over 32-key chunks.
// V fragments are loaded right after the score WMMAs so their latency is
// hidden under the softmax VALU work (the LDS barrier sits after them).
__global__ __launch_bounds__(128)
void attn_fwd(const bf16_t* __restrict__ Q, const bf16_t* __restrict__ K,
              const bf16_t* __restrict__ Vt, bf16_t* __restrict__ O) {
  __shared__ bf16_t ptile[4][16 * 32];          // per-wave P-transpose staging
  const int lane = threadIdx.x & 31;
  const int wid  = threadIdx.x >> 5;
  const int half = lane >> 4;
  const int lr   = lane & 15;
  const int b  = blockIdx.z;
  const int h  = blockIdx.y;
  const int qt = blockIdx.x * 4 + wid;
  const int q_base = qt * 16;

  // Q A-fragments (row = lr, dk along elements); dk 0-31 and 32-63
  const bf16_t* qptr = Q + (size_t)(b * S + q_base + lr) * D + h * DK + half * 8;
  const v16bf aq0 = load_frag(qptr);
  const v16bf aq1 = load_frag(qptr + 32);

  const bf16_t* kbaseptr = K + (size_t)(b * S) * D + h * DK + half * 8;
  const bf16_t* vbaseptr = Vt + ((size_t)(b * H + h) * DK) * S + half * 8;

  v8f acc[4] = {};
  float m[8], l[8];
#pragma unroll
  for (int j = 0; j < 8; ++j) { m[j] = -1e30f; l[j] = 0.0f; }

  bf16_t* myp = ptile[wid];
  const int nchunks = q_base / 32 + 1;          // keys 0 .. q_base+15 inclusive

  for (int kc = 0; kc < nchunks; ++kc) {
    const int kbase = kc * 32;

    // prefetch next chunk's K rows
    __builtin_prefetch(kbaseptr + (size_t)(kbase + 32 + lr) * D, 0, 1);

    // ---- scores S = (Q K^T) : two 16x16 key-groups, K-dim = DK in 2 steps
    v8f sc[2] = {};
    const bf16_t* kp = kbaseptr + (size_t)kbase * D;
#pragma unroll
    for (int g = 0; g < 2; ++g) {
      v16bf bk0 = load_frag(kp + (size_t)(g * 16 + lr) * D);        // dk 0-31
      v16bf bk1 = load_frag(kp + (size_t)(g * 16 + lr) * D + 32);   // dk 32-63
      sc[g] = wmma_bf16(aq0, bk0, sc[g]);
      sc[g] = wmma_bf16(aq1, bk1, sc[g]);
    }

    // ---- issue V fragment loads now; consumed after the softmax block
    const bf16_t* vp = vbaseptr + kbase;
    v16bf bv[4];
#pragma unroll
    for (int g = 0; g < 4; ++g) bv[g] = load_frag(vp + (size_t)(g * 16 + lr) * S);

    // ---- scale + causal mask
    const bool needmask = (kbase + 31) > q_base;
#pragma unroll
    for (int g = 0; g < 2; ++g) {
#pragma unroll
      for (int j = 0; j < 8; ++j) {
        float sv = sc[g][j] * 0.125f;           // 1/sqrt(DK)
        if (needmask) {
          int key = kbase + g * 16 + lr;
          int qq  = q_base + j + 8 * half;
          if (key > qq) sv = -1e30f;
        }
        sc[g][j] = sv;
      }
    }

    // ---- online softmax: row stats per (half-wave, VGPR j) row
    float alpha[8];
#pragma unroll
    for (int j = 0; j < 8; ++j) {
      float v = fmaxf(sc[0][j], sc[1][j]);
      v = fmaxf(v, __shfl_xor(v, 1, 32));
      v = fmaxf(v, __shfl_xor(v, 2, 32));
      v = fmaxf(v, __shfl_xor(v, 4, 32));
      v = fmaxf(v, __shfl_xor(v, 8, 32));
      float mn = fmaxf(m[j], v);
      alpha[j] = __expf(m[j] - mn);
      m[j] = mn;
    }
#pragma unroll
    for (int j = 0; j < 8; ++j) {
      float p0 = __expf(sc[0][j] - m[j]);
      float p1 = __expf(sc[1][j] - m[j]);
      float rs = p0 + p1;
      rs += __shfl_xor(rs, 1, 32);
      rs += __shfl_xor(rs, 2, 32);
      rs += __shfl_xor(rs, 4, 32);
      rs += __shfl_xor(rs, 8, 32);
      l[j] = l[j] * alpha[j] + rs;
      int row = j + 8 * half;                   // transpose P through LDS
      myp[row * 32 + lr]      = f2bf(p0);
      myp[row * 32 + 16 + lr] = f2bf(p1);
    }
#pragma unroll
    for (int g = 0; g < 4; ++g)
#pragma unroll
      for (int j = 0; j < 8; ++j) acc[g][j] *= alpha[j];

    asm volatile("s_wait_dscnt 0" ::: "memory");   // P visible in LDS
    v16bf ap = load_frag(myp + lr * 32 + half * 8);

    // ---- O += P * V  (B-frag: lane = dk column, keys along elements)
#pragma unroll
    for (int g = 0; g < 4; ++g) acc[g] = wmma_bf16(ap, bv[g], acc[g]);
  }

  // ---- normalize and store O (bf16, (B*S, H*DK) layout)
#pragma unroll
  for (int g = 0; g < 4; ++g) {
    int dk = g * 16 + lr;
#pragma unroll
    for (int j = 0; j < 8; ++j) {
      int row = q_base + j + 8 * half;
      float o = acc[g][j] / l[j];
      O[(size_t)(b * S + row) * D + h * DK + dk] = f2bf(o);
    }
  }
}

// ---------------------------------------------------------------- launcher
extern "C" void kernel_launch(void* const* d_in, const int* in_sizes, int n_in,
                              void* d_out, int out_size, void* d_ws, size_t ws_size,
                              hipStream_t stream) {
  (void)in_sizes; (void)n_in; (void)out_size; (void)ws_size;
  const float* x  = (const float*)d_in[0];
  // d_in[1] is the causal mask; causality is handled analytically.
  const float* wq = (const float*)d_in[2];
  const float* wk = (const float*)d_in[3];
  const float* wv = (const float*)d_in[4];
  const float* wo = (const float*)d_in[5];

  char* ws = (char*)d_ws;
  bf16_t* xb   = (bf16_t*)(ws);
  bf16_t* wqb  = (bf16_t*)(ws + (size_t)16 * (1u << 20));
  bf16_t* wkb  = (bf16_t*)(ws + (size_t)18 * (1u << 20));
  bf16_t* wvb  = (bf16_t*)(ws + (size_t)20 * (1u << 20));
  bf16_t* wob  = (bf16_t*)(ws + (size_t)22 * (1u << 20));
  float*  ftmp = (float*) (ws + (size_t)24 * (1u << 20));   // 32MB fp32 temp
  bf16_t* ob   = (bf16_t*)(ws + (size_t)24 * (1u << 20));   // reuse after V done
  bf16_t* qb   = (bf16_t*)(ws + (size_t)56 * (1u << 20));
  bf16_t* kb   = (bf16_t*)(ws + (size_t)72 * (1u << 20));
  bf16_t* vt   = (bf16_t*)(ws + (size_t)88 * (1u << 20));

  const int MD = M * D;   // 8,388,608
  const int DD = D * D;   // 1,048,576

  cvt_f32_bf16<<<(MD + 255) / 256, 256, 0, stream>>>(x,  xb,  MD);
  cvt_f32_bf16<<<(DD + 255) / 256, 256, 0, stream>>>(wq, wqb, DD);
  cvt_f32_bf16<<<(DD + 255) / 256, 256, 0, stream>>>(wk, wkb, DD);
  cvt_f32_bf16<<<(DD + 255) / 256, 256, 0, stream>>>(wv, wvb, DD);
  cvt_f32_bf16<<<(DD + 255) / 256, 256, 0, stream>>>(wo, wob, DD);

  dim3 gg(D / 256, M / 64);              // (4, 128) blocks of 256 threads
  const int rpN = M * H * (DK / 2);      // 4,194,304 rope pairs

  gemm_xwT<<<gg, 256, 0, stream>>>(xb, wqb, ftmp);
  rope_cvt<<<(rpN + 255) / 256, 256, 0, stream>>>(ftmp, qb);

  gemm_xwT<<<gg, 256, 0, stream>>>(xb, wkb, ftmp);
  rope_cvt<<<(rpN + 255) / 256, 256, 0, stream>>>(ftmp, kb);

  gemm_xwT<<<gg, 256, 0, stream>>>(xb, wvb, ftmp);
  vtrans<<<(MD + 255) / 256, 256, 0, stream>>>(ftmp, vt);

  attn_fwd<<<dim3(S / 64, H, Bsz), 128, 0, stream>>>(qb, kb, vt, ob);

  gemm_xwT<<<gg, 256, 0, stream>>>(ob, wob, (float*)d_out);
}
